// BinarySimpleCNN_22471268892741
// MI455X (gfx1250) — matrix operand order repaired and checked by reading, not compiled, not run on hardware
//
#include <hip/hip_runtime.h>

// ---------- CDNA5 WMMA types ----------
typedef __attribute__((ext_vector_type(16))) __bf16 v16bf;
typedef __attribute__((ext_vector_type(8)))  float  v8f;

// ---------- bf16 helpers (raw-bit storage as unsigned short) ----------
__device__ __forceinline__ unsigned short f32_to_bf16_bits(float f) {
  unsigned int u = __float_as_uint(f);
  u += 0x7FFFu + ((u >> 16) & 1u);     // round-to-nearest-even
  return (unsigned short)(u >> 16);
}
__device__ __forceinline__ __bf16 bits_to_bf16(unsigned short b) {
  return __builtin_bit_cast(__bf16, b);
}

// Fragment K-index maps, contiguous k = 16*half + e for A and B (16-bit A 16x32, B 32x16, wave32)
__device__ __forceinline__ int a_k_of(int e, int half) {
  return e + 16 * half;
}
__device__ __forceinline__ int b_k_of(int e, int half) { return e + 16 * half; }

// ---------- prep: fc weights, out[c][k] = bf16(sign(w[c][k])) ----------
__global__ void sign_fc_prep(const float* __restrict__ w,
                             unsigned short* __restrict__ o, int total) {
  int i = blockIdx.x * blockDim.x + threadIdx.x;
  if (i >= total) return;
  float x = w[i];
  float v = (x > 0.f) ? 1.f : ((x < 0.f) ? -1.f : 0.f);
  o[i] = f32_to_bf16_bits(v);
}

// ---------- prep: conv weights, reordered K = tap*CINP + cin, zero padded ----
// w layout [Cout][CIN][3][3]; o layout [Cout][KP] with k = t*CINP + cin
__global__ void sign_conv_prep(const float* __restrict__ w,
                               unsigned short* __restrict__ o,
                               int Cout, int CIN, int CINP, int KP) {
  int i = blockIdx.x * blockDim.x + threadIdx.x;
  if (i >= Cout * KP) return;
  int k = i % KP, c = i / KP;
  int cin = k % CINP, t = k / CINP;
  float v = 0.f;
  if (cin < CIN && t < 9) {
    float x = w[(c * CIN + cin) * 9 + t];
    v = (x > 0.f) ? 1.f : ((x < 0.f) ? -1.f : 0.f);
  }
  o[i] = f32_to_bf16_bits(v);
}

__global__ void zero_f32(float* __restrict__ p, int n) {
  int i = blockIdx.x * blockDim.x + threadIdx.x;
  if (i < n) p[i] = 0.f;
}

// ---------- fused binary conv3x3(pad1) + bias + relu + maxpool2x2 ----------
// 4 waves / workgroup. Tile: 16 cout x (8 rows x 8 cols); wave w owns rows
// 2w..2w+1 so 2x2 pooling closes in-wave. Input patch staged once in LDS,
// channel-INNERMOST layout [yy][xx][cin] so each lane's 16 B-fragment
// elements are contiguous shorts (merge into ds_load_b128 pairs).
// grid = (W/8, H/8, B * COUT/16), block = 128
template <int CIN, int CINP, int COUT, int H, int W, bool INF32>
__global__ __launch_bounds__(128) void bconv_relu_pool(
    const void* __restrict__ in_, const unsigned short* __restrict__ wsgn,
    const float* __restrict__ bias, unsigned short* __restrict__ out) {
  constexpr int K  = 9 * CINP;                 // reordered, cin zero-padded
  constexpr int KP = (K + 31) & ~31;
  constexpr int CT = COUT / 16;
  const float*          inf = (const float*)in_;
  const unsigned short* inb = (const unsigned short*)in_;

  const int tid  = threadIdx.x;
  const int lane = tid & 31, w = tid >> 5;
  const int n = lane & 15, half = lane >> 4;
  const int x0 = blockIdx.x * 8;
  const int y0 = blockIdx.y * 8;
  const int ct = blockIdx.z % CT;
  const int b  = blockIdx.z / CT;
  const int prow = n >> 3, pcol = n & 7;       // pixel within wave's 2x8 tile

  __shared__ unsigned short patch[100 * CINP]; // [(yy*10+xx)][cin]
  __shared__ float tile[4][16][16];

  // ---- stage input patch (rows y0-1..y0+8, cols x0-1..x0+8) ----
  for (int i = tid; i < 100 * CINP; i += 128) {
    int cin  = i & (CINP - 1);
    int yyxx = i / CINP;
    int yi = y0 - 1 + yyxx / 10;
    int xi = x0 - 1 + yyxx % 10;
    unsigned short v = 0;
    if (cin < CIN && yi >= 0 && yi < H && xi >= 0 && xi < W) {
      long idx = ((long)(b * CIN + cin) * H + yi) * W + xi;
      if constexpr (INF32) v = f32_to_bf16_bits(inf[idx]);
      else                 v = inb[idx];
    }
    patch[i] = v;
  }
  __syncthreads();

  // ---- implicit GEMM: D[16 cout][16 pix] over K ----
  v8f acc = {};
  for (int kc = 0; kc < KP; kc += 32) {
    v16bf Af, Bf;
#pragma unroll
    for (int e = 0; e < 16; ++e) {
      int ka = a_k_of(e, half) + kc;           // < KP (zero-padded weights)
      Af[e] = bits_to_bf16(wsgn[(ct * 16 + n) * KP + ka]);
    }
#pragma unroll
    for (int e = 0; e < 16; ++e) {
      int kb = kc + 16 * half + e;             // b_k_of
      int cin = kb & (CINP - 1);
      int t   = kb / CINP;                     // tap; const per chunk if CINP=32
      int dy = t / 3, dx = t % 3;
      unsigned short pv = 0;
      if (kb < K) {                            // guard only when KP > K
        int yy = 2 * w + prow + dy;            // 0..9
        int xx = pcol + dx;                    // 0..9
        pv = patch[(yy * 10 + xx) * CINP + cin];
      }
      Bf[e] = bits_to_bf16(pv);
    }
    acc = __builtin_amdgcn_wmma_f32_16x16x32_bf16(false, Af, false, Bf,
                                                  (short)0, acc, false, false);
  }

  // ---- bias + relu -> per-wave LDS tile, 2x2 pool -> bf16 out ----
#pragma unroll
  for (int r = 0; r < 8; ++r) {
    int m = r + 8 * half;
    float v = acc[r] + bias[ct * 16 + m];
    tile[w][m][n] = v > 0.f ? v : 0.f;
  }
  __syncthreads();

  constexpr int Hp = H / 2, Wp = W / 2;
#pragma unroll
  for (int t = 0; t < 2; ++t) {
    int o = lane + 32 * t;                     // 64 pooled outputs per wave
    int m = o >> 2, pc = o & 3;
    float a0 = tile[w][m][2 * pc],     a1 = tile[w][m][2 * pc + 1];
    float a2 = tile[w][m][8 + 2 * pc], a3 = tile[w][m][8 + 2 * pc + 1];
    float mx = fmaxf(fmaxf(a0, a1), fmaxf(a2, a3));
    long oidx = ((long)(b * COUT + ct * 16 + m) * Hp + (blockIdx.y * 4 + w)) * Wp +
                blockIdx.x * 4 + pc;
    out[oidx] = f32_to_bf16_bits(mx);
  }
}

// ---------- fc1: split-K WMMA, D[feat][batch] += A(sign wf1) x B(h3) ----------
// grid = (8 mtiles, 8 ntiles, 56 k-slices of 28 chunks); 50176 = 56*28*32
__global__ __launch_bounds__(32) void fc1_wmma(
    const unsigned short* __restrict__ h,     // [128 batch][50176] bf16
    const unsigned short* __restrict__ wf1s,  // [128 feat][50176] bf16
    float* __restrict__ acc) {                // [128 feat][128 batch] f32
  const int lane = threadIdx.x;
  const int n = lane & 15, half = lane >> 4;
  const int mt = blockIdx.x, nt = blockIdx.y;
  const int kbase = blockIdx.z * 28 * 32;

  v8f c = {};
  for (int j = 0; j < 28; ++j) {
    int kc = kbase + j * 32;
    v16bf Af, Bf;
#pragma unroll
    for (int e = 0; e < 16; ++e)
      Af[e] = bits_to_bf16(wf1s[(long)(mt * 16 + n) * 50176 + a_k_of(e, half) + kc]);
#pragma unroll
    for (int e = 0; e < 16; ++e)
      Bf[e] = bits_to_bf16(h[(long)(nt * 16 + n) * 50176 + b_k_of(e, half) + kc]);
    c = __builtin_amdgcn_wmma_f32_16x16x32_bf16(false, Af, false, Bf,
                                                (short)0, c, false, false);
  }
#pragma unroll
  for (int r = 0; r < 8; ++r) {
    int m = mt * 16 + r + 8 * half;
    atomicAdd(&acc[m * 128 + nt * 16 + n], c[r]);
  }
}

__global__ void fc1_finalize(const float* __restrict__ acc,
                             const float* __restrict__ bf1,
                             unsigned short* __restrict__ act) {
  int i = blockIdx.x * blockDim.x + threadIdx.x;
  if (i >= 128 * 128) return;
  int feat = i / 128;
  float v = acc[i] + bf1[feat];
  act[i] = f32_to_bf16_bits(v > 0.f ? v : 0.f);   // [feat][batch] bf16
}

// ---------- fc2: out[batch][1000] = fc1act @ sign(wf2).T + bf2 ----------
// grid = (63 mtiles covering 1008>=1000, 8 ntiles); K=128 -> 4 WMMA
__global__ __launch_bounds__(32) void fc2_wmma(
    const unsigned short* __restrict__ act,   // [128 feat][128 batch] bf16
    const unsigned short* __restrict__ wf2s,  // [1000][128] bf16
    const float* __restrict__ bf2, float* __restrict__ out) {
  const int lane = threadIdx.x;
  const int n = lane & 15, half = lane >> 4;
  const int mt = blockIdx.x, nt = blockIdx.y;
  const int feat = mt * 16 + n;

  v8f c = {};
#pragma unroll
  for (int kc = 0; kc < 128; kc += 32) {
    v16bf Af, Bf;
#pragma unroll
    for (int e = 0; e < 16; ++e) {
      int ka = a_k_of(e, half) + kc;
      unsigned short bb = (feat < 1000) ? wf2s[feat * 128 + ka] : (unsigned short)0;
      Af[e] = bits_to_bf16(bb);
    }
#pragma unroll
    for (int e = 0; e < 16; ++e) {
      int kb = b_k_of(e, half) + kc;
      Bf[e] = bits_to_bf16(act[kb * 128 + nt * 16 + n]);
    }
    c = __builtin_amdgcn_wmma_f32_16x16x32_bf16(false, Af, false, Bf,
                                                (short)0, c, false, false);
  }
#pragma unroll
  for (int r = 0; r < 8; ++r) {
    int m = mt * 16 + r + 8 * half;
    if (m < 1000) out[(long)(nt * 16 + n) * 1000 + m] = c[r] + bf2[m];
  }
}

// ---------- launch ----------
static inline size_t align256(size_t x) { return (x + 255) & ~(size_t)255; }

extern "C" void kernel_launch(void* const* d_in, const int* in_sizes, int n_in,
                              void* d_out, int out_size, void* d_ws, size_t ws_size,
                              hipStream_t stream) {
  (void)in_sizes; (void)n_in; (void)out_size; (void)ws_size;
  const float* x   = (const float*)d_in[0];
  const float* w1  = (const float*)d_in[1];
  const float* b1  = (const float*)d_in[2];
  const float* w2  = (const float*)d_in[3];
  const float* b2  = (const float*)d_in[4];
  const float* w3  = (const float*)d_in[5];
  const float* b3  = (const float*)d_in[6];
  const float* wf1 = (const float*)d_in[7];
  const float* bf1 = (const float*)d_in[8];
  const float* wf2 = (const float*)d_in[9];
  const float* bf2 = (const float*)d_in[10];
  float* out = (float*)d_out;

  // workspace layout (bf16 stored as unsigned short)
  char* base = (char*)d_ws;
  size_t off = 0;
  auto take = [&](size_t bytes) { char* p = base + off; off = align256(off + bytes); return p; };
  unsigned short* w1s  = (unsigned short*)take((size_t)16 * 64 * 2);    // KP=64 (CINP=4)
  unsigned short* w2s  = (unsigned short*)take((size_t)32 * 160 * 2);   // KP=160 (CINP=16)
  unsigned short* w3s  = (unsigned short*)take((size_t)64 * 288 * 2);   // KP=288 (CINP=32)
  unsigned short* wf1s = (unsigned short*)take((size_t)128 * 50176 * 2);
  unsigned short* wf2s = (unsigned short*)take((size_t)1000 * 128 * 2);
  unsigned short* h1   = (unsigned short*)take((size_t)128 * 16 * 112 * 112 * 2);
  unsigned short* h2   = (unsigned short*)take((size_t)128 * 32 * 56 * 56 * 2);
  unsigned short* h3   = (unsigned short*)take((size_t)128 * 64 * 28 * 28 * 2);
  float*          f1a  = (float*)take((size_t)128 * 128 * 4);
  unsigned short* f1x  = (unsigned short*)take((size_t)128 * 128 * 2);

  // 1) binarize weights to bf16 (conv: reordered K = tap*CINP + cin)
  sign_conv_prep<<<(16 * 64 + 255) / 256, 256, 0, stream>>>(w1, w1s, 16, 3, 4, 64);
  sign_conv_prep<<<(32 * 160 + 255) / 256, 256, 0, stream>>>(w2, w2s, 32, 16, 16, 160);
  sign_conv_prep<<<(64 * 288 + 255) / 256, 256, 0, stream>>>(w3, w3s, 64, 32, 32, 288);
  sign_fc_prep<<<(128 * 50176 + 255) / 256, 256, 0, stream>>>(wf1, wf1s, 128 * 50176);
  sign_fc_prep<<<(1000 * 128 + 255) / 256, 256, 0, stream>>>(wf2, wf2s, 1000 * 128);

  // 2) conv stack (LDS-staged implicit-GEMM WMMA, fused bias/relu/maxpool)
  bconv_relu_pool<3, 4, 16, 224, 224, true>
      <<<dim3(28, 28, 128 * 1), 128, 0, stream>>>(x, w1s, b1, h1);
  bconv_relu_pool<16, 16, 32, 112, 112, false>
      <<<dim3(14, 14, 128 * 2), 128, 0, stream>>>(h1, w2s, b2, h2);
  bconv_relu_pool<32, 32, 64, 56, 56, false>
      <<<dim3(7, 7, 128 * 4), 128, 0, stream>>>(h2, w3s, b3, h3);

  // 3) fc1 (split-K WMMA + f32 atomics), bias+relu, fc2
  zero_f32<<<(128 * 128 + 255) / 256, 256, 0, stream>>>(f1a, 128 * 128);
  fc1_wmma<<<dim3(8, 8, 56), 32, 0, stream>>>(h3, wf1s, f1a);
  fc1_finalize<<<(128 * 128 + 255) / 256, 256, 0, stream>>>(f1a, bf1, f1x);
  fc2_wmma<<<dim3(63, 8, 1), 32, 0, stream>>>(f1x, wf2s, bf2, out);
}
